// StackRNNModel_78280073937048
// MI455X (gfx1250) — matrix operand order repaired and checked by reading, not compiled, not run on hardware
//
#include <hip/hip_runtime.h>

#define SEQ    128
#define BATCH  32
#define NINP   1024
#define NTOKEN 32000
#define DEPTH  10
#define NG     (4 * NINP)
#define NLOG   (2 * (DEPTH + 1))

typedef _Float16 half_t;
typedef __attribute__((ext_vector_type(16))) _Float16 v16h;
typedef __attribute__((ext_vector_type(8)))  _Float16 v8h;
typedef __attribute__((ext_vector_type(4)))  _Float16 v4h;
typedef __attribute__((ext_vector_type(8)))  float    v8f;

union AF { v16h v; v8h h[2]; };

// ---------------------------------------------------------------------------
// f32 -> f16 conversion (weights / enc), 4 elements per thread
// ---------------------------------------------------------------------------
__global__ void k_cvt(const float* __restrict__ in, half_t* __restrict__ out, int n) {
  int i = (blockIdx.x * blockDim.x + threadIdx.x) * 4;
  if (i < n) {
    float4 v = *(const float4*)(in + i);
    v4h o = { (half_t)v.x, (half_t)v.y, (half_t)v.z, (half_t)v.w };
    *(v4h*)(out + i) = o;
  }
}

// ---------------------------------------------------------------------------
// Embedding gather straight to f16: emb_h[t,b,:] = (f16)enc[inputs[t,b],:]
// ---------------------------------------------------------------------------
__global__ void k_embed(const int* __restrict__ inputs,
                        const float* __restrict__ enc,
                        half_t* __restrict__ emb_h) {
  int row = blockIdx.x;                       // t*BATCH + b
  int tok = inputs[row];
  int i = threadIdx.x * 4;                    // 256 threads x 4 = 1024
  float4 v = *(const float4*)(enc + (size_t)tok * NINP + i);
  v4h o = { (half_t)v.x, (half_t)v.y, (half_t)v.z, (half_t)v.w };
  *(v4h*)(emb_h + (size_t)row * NINP + i) = o;
}

// ---------------------------------------------------------------------------
// Initial state into workspace (h also as f16 for the first gates GEMM)
// ---------------------------------------------------------------------------
__global__ void k_init(const float* __restrict__ h0, const float* __restrict__ c0,
                       const float* __restrict__ m0,
                       float* __restrict__ h, half_t* __restrict__ h_h,
                       float* __restrict__ c, float* __restrict__ mem) {
  int i = blockIdx.x * blockDim.x + threadIdx.x;
  if (i < BATCH * NINP) { h[i] = h0[i]; h_h[i] = (half_t)h0[i]; c[i] = c0[i]; }
  if (i < BATCH * DEPTH * NINP) mem[i] = m0[i];
}

// ---------------------------------------------------------------------------
// gates = x @ w_ih.T + h @ w_hh.T + b_ih + b_hh   (f16 WMMA, f32 accum)
// M=32, N=4096, K=1024 twice -> 64 wmma_16x16x32 per wave.
// Block = 128 = 4 waves (2 m-tiles x 2 n-tiles); grid.x = NG/32 = 128.
// ---------------------------------------------------------------------------
__global__ void __launch_bounds__(128)
k_gates(const half_t* __restrict__ x, const half_t* __restrict__ hh,
        const half_t* __restrict__ wih, const half_t* __restrict__ whh,
        const float* __restrict__ b_ih, const float* __restrict__ b_hh,
        float* __restrict__ gates) {
  int lane  = threadIdx.x & 31;
  int wave  = threadIdx.x >> 5;
  int mtile = wave & 1;
  int ntile = blockIdx.x * 2 + (wave >> 1);
  int mrow  = mtile * 16 + (lane & 15);
  int jcol  = ntile * 16 + (lane & 15);
  int ah    = (lane >> 4) * 8;
  int bh    = (lane >> 4) * 8;

  v8f acc = {};
  const half_t* arow = x + (size_t)mrow * NINP + ah;
  const half_t* brow = wih + (size_t)jcol * NINP + bh;
  for (int k0 = 0; k0 < NINP; k0 += 32) {
    AF a, b;
    a.h[0] = *(const v8h*)(arow + k0);
    a.h[1] = *(const v8h*)(arow + k0 + 16);
    b.h[0] = *(const v8h*)(brow + k0);
    b.h[1] = *(const v8h*)(brow + k0 + 16);
    acc = __builtin_amdgcn_wmma_f32_16x16x32_f16(false, a.v, false, b.v,
                                                 (short)0, acc, false, false);
  }
  arow = hh + (size_t)mrow * NINP + ah;
  brow = whh + (size_t)jcol * NINP + bh;
  for (int k0 = 0; k0 < NINP; k0 += 32) {
    AF a, b;
    a.h[0] = *(const v8h*)(arow + k0);
    a.h[1] = *(const v8h*)(arow + k0 + 16);
    b.h[0] = *(const v8h*)(brow + k0);
    b.h[1] = *(const v8h*)(brow + k0 + 16);
    acc = __builtin_amdgcn_wmma_f32_16x16x32_f16(false, a.v, false, b.v,
                                                 (short)0, acc, false, false);
  }

  float bias = b_ih[jcol] + b_hh[jcol];
  int mbase = mtile * 16 + ((lane >> 4) << 3);
#pragma unroll
  for (int r = 0; r < 8; ++r)
    gates[(size_t)(mbase + r) * NG + jcol] = acc[r] + bias;
}

// ---------------------------------------------------------------------------
// Fused per-batch step tail: LSTM cell -> policy softmax -> stack update.
// grid = BATCH (block owns batch b), block = 256 (each thread owns 4 n's).
// ---------------------------------------------------------------------------
__global__ void __launch_bounds__(256)
k_step2(const float* __restrict__ gates, const half_t* __restrict__ x,
        float* __restrict__ c, float* __restrict__ h, half_t* __restrict__ h_h,
        half_t* __restrict__ hs_t, float* __restrict__ mem,
        const float* __restrict__ conv_w, const float* __restrict__ conv_b,
        const float* __restrict__ pol_w, const float* __restrict__ pol_b) {
  __shared__ float red[8][NLOG];
  __shared__ float spol[NLOG];
  int b    = blockIdx.x;
  int tid  = threadIdx.x;
  int lane = tid & 31, wave = tid >> 5;

  const float*  g   = gates + (size_t)b * NG;
  const half_t* xb  = x + (size_t)b * NINP;
  float*        cb  = c + (size_t)b * NINP;
  float*        hb  = h + (size_t)b * NINP;
  half_t*       hhb = h_h + (size_t)b * NINP;
  half_t*       hsb = hs_t + (size_t)b * NINP;
  float*        mb  = mem + (size_t)b * DEPTH * NINP;

  float hn[4];
  float mvs[4][DEPTH];
  float part[NLOG];
#pragma unroll
  for (int l = 0; l < NLOG; ++l) part[l] = 0.f;

#pragma unroll
  for (int i = 0; i < 4; ++i) {
    int n = tid * 4 + i;
    float gi = g[n], gf = g[NINP + n], gg = g[2 * NINP + n], go = g[3 * NINP + n];
    float si = 1.f / (1.f + __expf(-gi));
    float sf = 1.f / (1.f + __expf(-gf));
    float so = 1.f / (1.f + __expf(-go));
    float cn = sf * cb[n] + si * tanhf(gg);
    float hv = so * tanhf(cn);
    cb[n] = cn; hb[n] = hv; hhb[n] = (half_t)hv; hsb[n] = (half_t)hv;
    hn[i] = hv;
#pragma unroll
    for (int j = 0; j < DEPTH; ++j) mvs[i][j] = mb[j * NINP + n];
    float xv = (float)xb[n];
    float cw00 = conv_w[(0 * NINP + n) * 2 + 0];
    float cw01 = conv_w[(0 * NINP + n) * 2 + 1];
    float cw10 = conv_w[(1 * NINP + n) * 2 + 0];
    float cw11 = conv_w[(1 * NINP + n) * 2 + 1];
#pragma unroll
    for (int l = 0; l < NLOG; ++l) {
      int o = l / (DEPTH + 1), tt = l % (DEPTH + 1);
      float a = xv * pol_w[(size_t)l * NINP + n];
      float c0 = o ? cw10 : cw00;
      float c1 = o ? cw11 : cw01;
      if (tt < DEPTH)     a += mvs[i][tt] * c0;
      if (tt + 1 < DEPTH) a += mvs[i][tt + 1] * c1;
      part[l] += a;
    }
  }

#pragma unroll
  for (int l = 0; l < NLOG; ++l) {
    float v = part[l];
#pragma unroll
    for (int off = 16; off > 0; off >>= 1) v += __shfl_down(v, off, 32);
    if (lane == 0) red[wave][l] = v;
  }
  __syncthreads();
  if (tid < NLOG) {
    float s = 0.f;
#pragma unroll
    for (int w = 0; w < 8; ++w) s += red[w][tid];
    red[0][tid] = s + conv_b[tid / (DEPTH + 1)] + pol_b[tid];
  }
  __syncthreads();
  if (tid == 0) {
    float mx = red[0][0];
    for (int i = 1; i < NLOG; ++i) mx = fmaxf(mx, red[0][i]);
    float e[NLOG]; float s = 0.f;
    for (int i = 0; i < NLOG; ++i) { e[i] = __expf(red[0][i] - mx); s += e[i]; }
    float inv = 1.f / s;
    for (int i = 0; i < NLOG; ++i) spol[i] = e[i] * inv;
  }
  __syncthreads();

#pragma unroll
  for (int i = 0; i < 4; ++i) {
    int n = tid * 4 + i;
#pragma unroll
    for (int d = 0; d < DEPTH; ++d) {
      float s = 0.f;
#pragma unroll
      for (int k = 0; k <= DEPTH; ++k) {
        int j = k + d;
        float stay = (j < DEPTH) ? mvs[i][j] : 0.f;
        float push = (d == 0) ? hn[i] : ((j - 1 < DEPTH) ? mvs[i][j - 1] : 0.f);
        s += spol[k] * stay + spol[DEPTH + 1 + k] * push;
      }
      mb[d * NINP + n] = s;
    }
  }
}

// ---------------------------------------------------------------------------
// Decoder GEMM: out[m,v] = sum_k hs[m,k]*enc[v,k] + dec_bias[v]
// f16 WMMA 16x16x32. B tiles (64 rows x 64 K, shared by all 8 waves) are
// staged into LDS once per block via async-to-LDS DMA (ASYNCcnt), double
// buffered; waves read fragments with ds_load_b128. A frags direct global.
// grid = (2000/4, 256/8), block = 256 = 8 waves. LDS = 2*64*72*2B = 18 KB.
// ---------------------------------------------------------------------------
#define BK    64
#define BPAD  8                       // row stride 72 halves -> conflict-free
__global__ void __launch_bounds__(256)
k_decoder(const half_t* __restrict__ hs, const half_t* __restrict__ enc,
          const float* __restrict__ dec_bias, float* __restrict__ out) {
  __shared__ half_t Bs[2][64][BK + BPAD];
  int tid    = threadIdx.x;
  int lane   = tid & 31;
  int wave   = tid >> 5;
  int mtile  = blockIdx.y * 8 + wave;
  int n0base = blockIdx.x * 4 * 16;
  int mrow   = mtile * 16 + (lane & 15);
  int ah     = (lane >> 4) * 8;
  int bh     = (lane >> 4) * 8;

  const half_t* arow = hs + (size_t)mrow * NINP + ah;

  // this thread's two 16B async-copy chunks per staged block
  int id0 = tid * 2, id1 = tid * 2 + 1;          // 0..511 over 64 rows x 8 chunks
  int r0 = id0 >> 3, c0 = (id0 & 7) * 8;
  int r1 = id1 >> 3, c1 = (id1 & 7) * 8;
  const half_t* g0 = enc + (size_t)(n0base + r0) * NINP + c0;
  const half_t* g1 = enc + (size_t)(n0base + r1) * NINP + c1;

#define ASYNC_STAGE(buf, kb)                                                   \
  do {                                                                         \
    unsigned _l0 = (unsigned)(size_t)&Bs[buf][r0][c0];                         \
    unsigned _l1 = (unsigned)(size_t)&Bs[buf][r1][c1];                         \
    asm volatile("global_load_async_to_lds_b128 %0, %1, off"                   \
                 :: "v"(_l0), "v"(g0 + (kb)) : "memory");                      \
    asm volatile("global_load_async_to_lds_b128 %0, %1, off"                   \
                 :: "v"(_l1), "v"(g1 + (kb)) : "memory");                      \
  } while (0)

  v8f acc[4] = {{}, {}, {}, {}};

  ASYNC_STAGE(0, 0);
  asm volatile("s_wait_asynccnt 0x0" ::: "memory");
  __syncthreads();

  int buf = 0;
  for (int blk = 0; blk < NINP / BK; ++blk) {
    if (blk + 1 < NINP / BK) ASYNC_STAGE(buf ^ 1, (blk + 1) * BK);
#pragma unroll
    for (int s = 0; s < 2; ++s) {                 // two K=32 steps per block
      int k0g = blk * BK + s * 32;
      AF a;
      a.h[0] = *(const v8h*)(arow + k0g);
      a.h[1] = *(const v8h*)(arow + k0g + 16);
#pragma unroll
      for (int j = 0; j < 4; ++j) {
        const half_t* lp = &Bs[buf][j * 16 + (lane & 15)][s * 32 + bh];
        AF bb;
        bb.h[0] = *(const v8h*)lp;
        bb.h[1] = *(const v8h*)(lp + 16);
        acc[j] = __builtin_amdgcn_wmma_f32_16x16x32_f16(false, a.v, false, bb.v,
                                                        (short)0, acc[j], false, false);
      }
    }
    asm volatile("s_wait_asynccnt 0x0" ::: "memory");
    __syncthreads();
    buf ^= 1;
  }
#undef ASYNC_STAGE

  int mbase = mtile * 16 + ((lane >> 4) << 3);
#pragma unroll
  for (int j = 0; j < 4; ++j) {
    int n = n0base + j * 16 + (lane & 15);
    float bias = dec_bias[n];
#pragma unroll
    for (int r = 0; r < 8; ++r)
      out[(size_t)(mbase + r) * NTOKEN + n] = acc[j][r] + bias;
  }
}

// ---------------------------------------------------------------------------
// Tail outputs: hT, cT, memT after the decoded block
// ---------------------------------------------------------------------------
__global__ void k_final(const float* __restrict__ h, const float* __restrict__ c,
                        const float* __restrict__ mem, float* __restrict__ out) {
  const size_t base = (size_t)SEQ * BATCH * NTOKEN;
  int i = blockIdx.x * blockDim.x + threadIdx.x;
  const int HN = BATCH * NINP;
  const int MN = BATCH * DEPTH * NINP;
  if (i < HN)               out[base + i] = h[i];
  else if (i < 2 * HN)      out[base + i] = c[i - HN];
  else if (i < 2 * HN + MN) out[base + i] = mem[i - 2 * HN];
}

// ---------------------------------------------------------------------------
extern "C" void kernel_launch(void* const* d_in, const int* in_sizes, int n_in,
                              void* d_out, int out_size, void* d_ws, size_t ws_size,
                              hipStream_t stream) {
  (void)in_sizes; (void)n_in; (void)out_size; (void)ws_size;
  const int*   inputs = (const int*)  d_in[0];
  const float* h0     = (const float*)d_in[1];
  const float* c0     = (const float*)d_in[2];
  const float* mem0   = (const float*)d_in[3];
  const float* enc    = (const float*)d_in[4];
  const float* dec_b  = (const float*)d_in[5];
  const float* w_ih   = (const float*)d_in[6];
  const float* w_hh   = (const float*)d_in[7];
  const float* b_ih   = (const float*)d_in[8];
  const float* b_hh   = (const float*)d_in[9];
  const float* conv_w = (const float*)d_in[10];
  const float* conv_b = (const float*)d_in[11];
  const float* pol_w  = (const float*)d_in[12];
  const float* pol_b  = (const float*)d_in[13];
  float* out = (float*)d_out;

  // ---- workspace carve-out (~101 MB) ----
  char* P = (char*)d_ws;
  half_t* enc_h = (half_t*)P; P += (size_t)NTOKEN * NINP * sizeof(half_t);
  half_t* wih_h = (half_t*)P; P += (size_t)NG * NINP * sizeof(half_t);
  half_t* whh_h = (half_t*)P; P += (size_t)NG * NINP * sizeof(half_t);
  half_t* emb_h = (half_t*)P; P += (size_t)SEQ * BATCH * NINP * sizeof(half_t);
  half_t* hs_h  = (half_t*)P; P += (size_t)SEQ * BATCH * NINP * sizeof(half_t);
  half_t* h_h   = (half_t*)P; P += (size_t)BATCH * NINP * sizeof(half_t);
  float*  gates = (float*)P;  P += (size_t)BATCH * NG * sizeof(float);
  float*  h     = (float*)P;  P += (size_t)BATCH * NINP * sizeof(float);
  float*  c     = (float*)P;  P += (size_t)BATCH * NINP * sizeof(float);
  float*  mem   = (float*)P;  P += (size_t)BATCH * DEPTH * NINP * sizeof(float);

  // ---- one-time f32 -> f16 conversions ----
  k_cvt<<<(NTOKEN * NINP / 4 + 255) / 256, 256, 0, stream>>>(enc, enc_h, NTOKEN * NINP);
  k_cvt<<<(NG * NINP / 4 + 255) / 256, 256, 0, stream>>>(w_ih, wih_h, NG * NINP);
  k_cvt<<<(NG * NINP / 4 + 255) / 256, 256, 0, stream>>>(w_hh, whh_h, NG * NINP);

  k_init<<<1280, 256, 0, stream>>>(h0, c0, mem0, h, h_h, c, mem);
  k_embed<<<SEQ * BATCH, 256, 0, stream>>>(inputs, enc, emb_h);

  // ---- recurrent scan: 2 launches / step ----
  for (int t = 0; t < SEQ; ++t) {
    const half_t* x = emb_h + (size_t)t * BATCH * NINP;
    k_gates<<<NG / 32, 128, 0, stream>>>(x, h_h, wih_h, whh_h, b_ih, b_hh, gates);
    k_step2<<<BATCH, 256, 0, stream>>>(gates, x, c, h, h_h,
                                       hs_h + (size_t)t * BATCH * NINP, mem,
                                       conv_w, conv_b, pol_w, pol_b);
  }

  // ---- decoder GEMM + tail state ----
  k_decoder<<<dim3((NTOKEN / 16) / 4, (SEQ * BATCH / 16) / 8), 256, 0, stream>>>(
      hs_h, enc_h, dec_b, out);
  k_final<<<(2 * BATCH * NINP + BATCH * DEPTH * NINP + 255) / 256, 256, 0, stream>>>(
      h, c, mem, out);
}
